// LinearAttention_86921548136922
// MI455X (gfx1250) — matrix operand order repaired and checked by reading, not compiled, not run on hardware
//
#include <hip/hip_runtime.h>
#include <hip/hip_bf16.h>

// ---------------------------------------------------------------------------
// Linear attention (phi(x)=elu(x)+1) for MI455X / gfx1250 (wave32, WMMA).
// B=4, N=4096, D=E=1024.
//   Qp  = bf16(phi(Q))            [B,N,D]   (K-major for GEMM2 A)
//   KpT = bf16(phi(K))^T          [B,D,N]   (K-major)
//   VbT = bf16(V)^T               [B,E,N]   (K-major)
//   Ksum[b,d] = sum_n phi(K)      [B,D]
//   Z[b,n]    = Qp . Ksum + eps   [B,N]
//   KVT = bf16(VbT * KpT^T)  (== (Kp^T V)^T)  [B,E,D]  bf16, K-major for GEMM2 B
//   out = (Qp * KVT^T) / Z                    [B,N,E]  f32
// ---------------------------------------------------------------------------

typedef __attribute__((ext_vector_type(16))) __bf16          v16bf;
typedef __attribute__((ext_vector_type(8)))  float           v8f;
typedef __attribute__((ext_vector_type(16))) unsigned short  v16us;
typedef __attribute__((ext_vector_type(8)))  unsigned short  v8us;
typedef __attribute__((ext_vector_type(4)))  unsigned short  v4us;

union BF16Frag { v16us u; v16bf b; };

__device__ __forceinline__ float phi_f(float x) {
    // elu(x)+1 == (x>0) ? x+1 : exp(x)
    return x > 0.0f ? x + 1.0f : __expf(x);
}

__device__ __forceinline__ unsigned short f2bf(float f) {
    unsigned int u = __float_as_uint(f);
    u += 0x7FFFu + ((u >> 16) & 1u);      // round-to-nearest-even
    return (unsigned short)(u >> 16);
}

__device__ __forceinline__ float bf2f(unsigned short h) {
    return __uint_as_float(((unsigned int)h) << 16);
}

// ---------------------------------------------------------------------------
// Elementwise: Qp = bf16(phi(Q)), same layout.
// ---------------------------------------------------------------------------
__global__ __launch_bounds__(256)
void linattn_qprep(const float* __restrict__ Q, unsigned short* __restrict__ Qp) {
    const int i = (blockIdx.x * 256 + threadIdx.x) * 4;
    const float4 q = *(const float4*)(Q + i);
    v4us o;
    o[0] = f2bf(phi_f(q.x));
    o[1] = f2bf(phi_f(q.y));
    o[2] = f2bf(phi_f(q.z));
    o[3] = f2bf(phi_f(q.w));
    *(v4us*)(Qp + i) = o;
}

// ---------------------------------------------------------------------------
// LDS-tiled transpose + optional phi + f32->bf16.
// src [Nrows][Dcols] f32  ->  dst [Dcols][Nrows] bf16 (as ushort)
// 64x64 tile per 256-thread block.
// ---------------------------------------------------------------------------
template<bool PHI>
__global__ __launch_bounds__(256)
void linattn_transpose(const float* __restrict__ src, unsigned short* __restrict__ dst,
                       int Nrows, int Dcols) {
    __shared__ unsigned short tile[64][66];
    const int b = blockIdx.z;
    src += (size_t)b * Nrows * Dcols;
    dst += (size_t)b * Nrows * Dcols;
    const int d0 = blockIdx.x * 64;
    const int n0 = blockIdx.y * 64;
    const int tid = threadIdx.x;
    const int cx = tid & 63;      // column (d) on read, column (n) on write
    const int ry = tid >> 6;      // 0..3
#pragma unroll
    for (int i = 0; i < 16; ++i) {
        const int r = ry + i * 4;                       // n within tile
        float v = src[(size_t)(n0 + r) * Dcols + d0 + cx];
        if (PHI) v = phi_f(v);
        tile[r][cx] = f2bf(v);
    }
    __syncthreads();
#pragma unroll
    for (int i = 0; i < 16; ++i) {
        const int c = ry + i * 4;                       // d within tile
        dst[(size_t)(d0 + c) * Nrows + n0 + cx] = tile[cx][c];
    }
}

// ---------------------------------------------------------------------------
// Ksum[b,d] = sum_n phi(K[b,n,d]); coalesced across d.
// ---------------------------------------------------------------------------
__global__ __launch_bounds__(256)
void linattn_ksum(const float* __restrict__ K, float* __restrict__ Ksum,
                  int N, int D) {
    const int b = blockIdx.y;
    const int d = blockIdx.x * 256 + threadIdx.x;
    const float* Kb = K + (size_t)b * N * D;
    float acc = 0.0f;
    for (int n = 0; n < N; ++n)
        acc += phi_f(Kb[(size_t)n * D + d]);
    Ksum[b * D + d] = acc;
}

// ---------------------------------------------------------------------------
// Z[row] = Qp[row,:] . Ksum[b,:] + eps      (one wave32 per row)
// ---------------------------------------------------------------------------
__global__ __launch_bounds__(256)
void linattn_z(const unsigned short* __restrict__ Qp, const float* __restrict__ Ksum,
               float* __restrict__ Z, int N, int D) {
    const int row  = blockIdx.x * 8 + (threadIdx.x >> 5);
    const int lane = threadIdx.x & 31;
    const int b = row / N;
    const unsigned short* q = Qp + (size_t)row * D;
    const float* ks = Ksum + (size_t)b * D;
    float acc = 0.0f;
    for (int d = lane; d < D; d += 32)
        acc += bf2f(q[d]) * ks[d];
#pragma unroll
    for (int off = 16; off > 0; off >>= 1)
        acc += __shfl_xor(acc, off, 32);
    if (lane == 0) Z[row] = acc + 1e-6f;
}

// ---------------------------------------------------------------------------
// Batched GEMM: C[m][n] = sum_k A[m][k] * Bt[n][k]   (both operands K-major)
// bf16 inputs, f32 accumulate via v_wmma_f32_16x16x32_bf16.
// 256 threads = 8 waves; block tile 128x128; wave tile 64x32 (4x2 WMMA tiles).
// DIVZ: divide by Z[b*strideZ + m] on store (GEMM2 epilogue).
// OutT: float (store f32) or unsigned short (store bf16, RNE).
// ---------------------------------------------------------------------------
template<bool DIVZ, typename OutT>
__global__ __launch_bounds__(256)
void linattn_gemm(const unsigned short* __restrict__ A,
                  const unsigned short* __restrict__ Bt,
                  OutT* __restrict__ C,
                  const float* __restrict__ Zv,
                  int Ncols, int Kdim,
                  long strideA, long strideB, long strideC, long strideZ) {
    const int b = blockIdx.z;
    A  += (size_t)b * strideA;
    Bt += (size_t)b * strideB;
    C  += (size_t)b * strideC;

    const int tid  = threadIdx.x;
    const int wave = tid >> 5;
    const int lane = tid & 31;
    const int l16  = lane & 15;
    const int hlf  = lane >> 4;        // 0: lanes 0-15, 1: lanes 16-31

    const int wm = wave >> 2;          // 0..1
    const int wn = wave & 3;           // 0..3
    const int m0 = blockIdx.x * 128 + wm * 64;
    const int n0 = blockIdx.y * 128 + wn * 32;

    v8f acc[4][2];
#pragma unroll
    for (int tm = 0; tm < 4; ++tm)
#pragma unroll
        for (int tn = 0; tn < 2; ++tn)
            acc[tm][tn] = (v8f)(0.0f);

    const unsigned short* Arow[4];
#pragma unroll
    for (int tm = 0; tm < 4; ++tm)
        Arow[tm] = A + (size_t)(m0 + tm * 16 + l16) * (size_t)Kdim;
    const unsigned short* Brow[2];
#pragma unroll
    for (int tn = 0; tn < 2; ++tn)
        Brow[tn] = Bt + (size_t)(n0 + tn * 16 + l16) * (size_t)Kdim;

    for (int k = 0; k < Kdim; k += 32) {
        BF16Frag a[4], bb[2];
#pragma unroll
        for (int tm = 0; tm < 4; ++tm) {
            // A fragment (16-bit 16x32): lane holds M=l16; K = [k + hlf*16, +16)
            a[tm].u = *(const v16us*)(Arow[tm] + k + hlf * 16);
        }
#pragma unroll
        for (int tn = 0; tn < 2; ++tn) {
            // B fragment (32x16): lane holds N=l16; K = [k + hlf*16, +16)
            bb[tn].u = *(const v16us*)(Brow[tn] + k + hlf * 16);
        }
        // prefetch next K slice (speculative; OOB prefetch is dropped)
        __builtin_prefetch((const void*)(Arow[0] + k + 32 + hlf * 8), 0, 1);
        __builtin_prefetch((const void*)(Arow[2] + k + 32 + hlf * 8), 0, 1);
        __builtin_prefetch((const void*)(Brow[0] + k + 32 + hlf * 16), 0, 1);
        __builtin_prefetch((const void*)(Brow[1] + k + 32 + hlf * 16), 0, 1);
#pragma unroll
        for (int tm = 0; tm < 4; ++tm)
#pragma unroll
            for (int tn = 0; tn < 2; ++tn)
                acc[tm][tn] = __builtin_amdgcn_wmma_f32_16x16x32_bf16(
                    false, a[tm].b, false, bb[tn].b,
                    (short)0, acc[tm][tn], false, false);
    }

    // Epilogue: D fragment layout: VGPR i -> M = i + hlf*8, N = l16.
#pragma unroll
    for (int tm = 0; tm < 4; ++tm) {
        const int mbase = m0 + tm * 16;
#pragma unroll
        for (int tn = 0; tn < 2; ++tn) {
            const int cc = n0 + tn * 16 + l16;
#pragma unroll
            for (int i = 0; i < 8; ++i) {
                const int r = mbase + i + hlf * 8;
                float v = acc[tm][tn][i];
                if (DIVZ) v /= Zv[(size_t)b * strideZ + r];
                if constexpr (sizeof(OutT) == 2) {
                    C[(size_t)r * Ncols + cc] = (OutT)f2bf(v);
                } else {
                    C[(size_t)r * Ncols + cc] = (OutT)v;
                }
            }
        }
    }
}

// ---------------------------------------------------------------------------
extern "C" void kernel_launch(void* const* d_in, const int* in_sizes, int n_in,
                              void* d_out, int out_size, void* d_ws, size_t ws_size,
                              hipStream_t stream) {
    const int B = 4, N = 4096, D = 1024;
    const float* Q = (const float*)d_in[0];
    const float* K = (const float*)d_in[1];
    const float* V = (const float*)d_in[2];
    float* out = (float*)d_out;

    char* ws = (char*)d_ws;
    const size_t elems = (size_t)B * N * D;               // 16 Mi elements
    unsigned short* Qp  = (unsigned short*)ws;  ws += elems * sizeof(unsigned short);
    unsigned short* KpT = (unsigned short*)ws;  ws += elems * sizeof(unsigned short);
    unsigned short* VbT = (unsigned short*)ws;  ws += elems * sizeof(unsigned short);
    unsigned short* KVT = (unsigned short*)ws;  ws += (size_t)B * D * D * sizeof(unsigned short);
    float* Ksum = (float*)ws;                   ws += (size_t)B * D * sizeof(float);
    float* Zb   = (float*)ws;

    // Stage 0: elementwise / transpose prep
    linattn_qprep<<<(unsigned)(elems / (256 * 4)), 256, 0, stream>>>(Q, Qp);
    linattn_transpose<true ><<<dim3(D / 64, N / 64, B), 256, 0, stream>>>(K, KpT, N, D);
    linattn_transpose<false><<<dim3(D / 64, N / 64, B), 256, 0, stream>>>(V, VbT, N, D);
    linattn_ksum<<<dim3(D / 256, B), 256, 0, stream>>>(K, Ksum, N, D);
    linattn_z<<<(B * N) / 8, 256, 0, stream>>>(Qp, Ksum, Zb, N, D);

    // GEMM1: KVT[e][d] = bf16( sum_n VbT[e][n] * KpT[d][n] )   (M=E, Ncols=D, K=N)
    linattn_gemm<false, unsigned short><<<dim3(D / 128, D / 128, B), 256, 0, stream>>>(
        VbT, KpT, KVT, nullptr,
        D, N, (long)D * N, (long)D * N, (long)D * D, 0L);

    // GEMM2: out[n][e] = (sum_d Qp[n][d] * KVT[e][d]) / Z[n]   (M=N, Ncols=E, K=D)
    linattn_gemm<true, float><<<dim3(N / 128, D / 128, B), 256, 0, stream>>>(
        Qp, KVT, out, Zb,
        D, D, (long)N * D, (long)D * D, (long)N * D, (long)N);
}
